// GCN_11647951307437
// MI455X (gfx1250) — matrix operand order repaired and checked by reading, not compiled, not run on hardware
//
#include <hip/hip_runtime.h>
#include <hip/hip_bf16.h>
#include <math.h>

// ---------------------------------------------------------------------------
// GCN (2-layer) for MI455X / gfx1250, wave32, WMMA bf16 16x16x32 (f32 acc).
// ---------------------------------------------------------------------------

typedef __attribute__((ext_vector_type(16))) __bf16 v16bf;
typedef __attribute__((ext_vector_type(8)))  float  v8f;

#define N_NODES 100000
#define N_EDGES 1600000
#define D_IN    256
#define D_HID   128
#define D_OUT   40
#define D_OUT_P 48              // padded to 3 n-tiles of 16
#define M_TILES 6250            // 100000 / 16 exactly

// ---------------------------------------------------------------------------
// Prep: transpose weights, convert to bf16 (B-fragment wants contiguous K per
// fixed N column).  w1t[n][k] (128x256), w2t[n][k] (48x128, cols>=40 zero).
// ---------------------------------------------------------------------------
__global__ __launch_bounds__(256) void gcn_prep_kernel(
    const float* __restrict__ W1, const float* __restrict__ W2,
    __bf16* __restrict__ w1t, __bf16* __restrict__ w2t)
{
    int tid = blockIdx.x * 256 + threadIdx.x;
    if (tid < D_HID * D_IN) {                       // 32768
        int n = tid / D_IN, k = tid % D_IN;
        w1t[tid] = (__bf16)W1[k * D_HID + n];
    } else if (tid < D_HID * D_IN + D_OUT_P * D_HID) {
        int t = tid - D_HID * D_IN;                 // 6144
        int n = t / D_HID, k = t % D_HID;
        w2t[t] = (__bf16)((n < D_OUT) ? W2[k * D_OUT + n] : 0.0f);
    }
}

// ---------------------------------------------------------------------------
// GEMM1: support1[100000,128] = x[100000,256] @ W1.  One wave per 16-row
// M-tile; each wave keeps all 8 N-tile accumulators (reads x exactly once).
// W1t (bf16, 64KB) staged in LDS.
// ---------------------------------------------------------------------------
__global__ __launch_bounds__(256) void gcn_gemm1_kernel(
    const float* __restrict__ x, const __bf16* __restrict__ w1t,
    float* __restrict__ support1)
{
    __shared__ uint4 sraw[4096];                    // 64KB = 128*256 bf16
    __bf16* sW = (__bf16*)sraw;
    const uint4* g = (const uint4*)w1t;
    for (int i = threadIdx.x; i < 4096; i += 256) sraw[i] = g[i];
    __syncthreads();

    const int wave  = threadIdx.x >> 5;
    const int lane  = threadIdx.x & 31;
    const int mtile = blockIdx.x * 8 + wave;
    if (mtile >= M_TILES) return;

    const int m  = lane & 15;
    const int kA = (lane >> 4) * 16;                // A half-wave K split
    const int n  = lane & 15;
    const int kB = (lane >> 4) * 16;                // B half-wave K split

    const float* arow = x + (size_t)(mtile * 16 + m) * D_IN;

    v8f acc[8] = {};
    for (int k = 0; k < D_IN; k += 32) {
        v16bf a;
        const float* p = arow + k + kA;
#pragma unroll
        for (int i = 0; i < 8; ++i) a[i]     = (__bf16)p[i];        // K 0..7 / 16..23
#pragma unroll
        for (int i = 0; i < 8; ++i) a[8 + i] = (__bf16)p[8 + i];    // K 8..15 / 24..31
#pragma unroll
        for (int t = 0; t < 8; ++t) {
            v16bf b = *(const v16bf*)(sW + (t * 16 + n) * D_IN + k + kB);
            acc[t] = __builtin_amdgcn_wmma_f32_16x16x32_bf16(
                false, a, false, b, (short)0, acc[t], false, false);
        }
    }

    const int mb = (lane >> 4) * 8;                 // C/D: vgpr r -> row r / r+8
    float* orow = support1 + (size_t)(mtile * 16 + mb) * D_HID + n;
#pragma unroll
    for (int t = 0; t < 8; ++t)
#pragma unroll
        for (int r = 0; r < 8; ++r)
            orow[(size_t)r * D_HID + t * 16] = acc[t][r];
}

// ---------------------------------------------------------------------------
// agg1 = 0
// ---------------------------------------------------------------------------
__global__ __launch_bounds__(256) void gcn_zero_kernel(float4* __restrict__ p, int n4)
{
    int i = blockIdx.x * 256 + threadIdx.x;
    if (i < n4) p[i] = float4{0.f, 0.f, 0.f, 0.f};
}

// ---------------------------------------------------------------------------
// Scatter layer 1: wave per edge, 4 f32 atomics per lane (D=128).
// All traffic is L2-resident (support1/agg1 = 51.2MB each < 192MB L2).
// ---------------------------------------------------------------------------
__global__ __launch_bounds__(256) void gcn_scatter1_kernel(
    const float* __restrict__ support1, const int* __restrict__ esrc,
    const int* __restrict__ edst, const float* __restrict__ ew,
    float* __restrict__ agg1)
{
    int gid = blockIdx.x * 256 + threadIdx.x;
    int e = gid >> 5, lane = gid & 31;
    if (e >= N_EDGES) return;
    int s = esrc[e], d = edst[e];
    float w = ew[e];
    const float* sp = support1 + (size_t)s * D_HID;
    float*       ap = agg1     + (size_t)d * D_HID;
#pragma unroll
    for (int i = 0; i < 4; ++i) {
        int f = lane + 32 * i;
        atomicAdd(ap + f, sp[f] * w);
    }
}

// ---------------------------------------------------------------------------
// GEMM2: support2[100000,40] = relu(agg1 + b1) @ W2, relu+bias fused into the
// A-fragment load.  N padded to 48 (3 tiles), masked store of cols 0..39.
// ---------------------------------------------------------------------------
__global__ __launch_bounds__(256) void gcn_gemm2_kernel(
    const float* __restrict__ agg1, const float* __restrict__ b1,
    const __bf16* __restrict__ w2t, float* __restrict__ support2)
{
    __shared__ uint4 sraw[768];                     // 48*128 bf16 = 12288B
    __shared__ float sb1[D_HID];
    __bf16* sW = (__bf16*)sraw;
    const uint4* g = (const uint4*)w2t;
    for (int i = threadIdx.x; i < 768; i += 256) sraw[i] = g[i];
    if (threadIdx.x < D_HID) sb1[threadIdx.x] = b1[threadIdx.x];
    __syncthreads();

    const int wave  = threadIdx.x >> 5;
    const int lane  = threadIdx.x & 31;
    const int mtile = blockIdx.x * 8 + wave;
    if (mtile >= M_TILES) return;

    const int m  = lane & 15;
    const int kA = (lane >> 4) * 16;
    const int n  = lane & 15;
    const int kB = (lane >> 4) * 16;

    const float* arow = agg1 + (size_t)(mtile * 16 + m) * D_HID;

    v8f acc[3] = {};
    for (int k = 0; k < D_HID; k += 32) {
        v16bf a;
#pragma unroll
        for (int i = 0; i < 8; ++i) {
            float v = arow[k + kA + i] + sb1[k + kA + i];
            a[i] = (__bf16)fmaxf(v, 0.0f);
        }
#pragma unroll
        for (int i = 0; i < 8; ++i) {
            float v = arow[k + kA + 8 + i] + sb1[k + kA + 8 + i];
            a[8 + i] = (__bf16)fmaxf(v, 0.0f);
        }
#pragma unroll
        for (int t = 0; t < 3; ++t) {
            v16bf b = *(const v16bf*)(sW + (t * 16 + n) * D_HID + k + kB);
            acc[t] = __builtin_amdgcn_wmma_f32_16x16x32_bf16(
                false, a, false, b, (short)0, acc[t], false, false);
        }
    }

    const int mb = (lane >> 4) * 8;
#pragma unroll
    for (int t = 0; t < 3; ++t) {
        int col = t * 16 + n;
        if (col < D_OUT) {
            float* orow = support2 + (size_t)(mtile * 16 + mb) * D_OUT + col;
#pragma unroll
            for (int r = 0; r < 8; ++r) orow[(size_t)r * D_OUT] = acc[t][r];
        }
    }
}

// ---------------------------------------------------------------------------
// agg2[n][j] = b2[j]  (bias pre-seeded so scatter2 accumulates on top)
// ---------------------------------------------------------------------------
__global__ __launch_bounds__(256) void gcn_initb2_kernel(
    float* __restrict__ agg2, const float* __restrict__ b2)
{
    int i = blockIdx.x * 256 + threadIdx.x;
    if (i < N_NODES * D_OUT) agg2[i] = b2[i % D_OUT];
}

// ---------------------------------------------------------------------------
// Scatter layer 2: wave per edge, D=40 -> lanes cover f=lane and f=lane+32.
// ---------------------------------------------------------------------------
__global__ __launch_bounds__(256) void gcn_scatter2_kernel(
    const float* __restrict__ support2, const int* __restrict__ esrc,
    const int* __restrict__ edst, const float* __restrict__ ew,
    float* __restrict__ agg2)
{
    int gid = blockIdx.x * 256 + threadIdx.x;
    int e = gid >> 5, lane = gid & 31;
    if (e >= N_EDGES) return;
    int s = esrc[e], d = edst[e];
    float w = ew[e];
    const float* sp = support2 + (size_t)s * D_OUT;
    float*       ap = agg2     + (size_t)d * D_OUT;
    atomicAdd(ap + lane, sp[lane] * w);
    if (lane < D_OUT - 32) atomicAdd(ap + 32 + lane, sp[32 + lane] * w);
}

// ---------------------------------------------------------------------------
// log_softmax over 40 classes, wave per node, __shfl_xor wave32 reductions.
// ---------------------------------------------------------------------------
__global__ __launch_bounds__(256) void gcn_lsm_kernel(
    const float* __restrict__ agg2, float* __restrict__ out)
{
    int gid = blockIdx.x * 256 + threadIdx.x;
    int node = gid >> 5, lane = gid & 31;
    if (node >= N_NODES) return;
    const float* r = agg2 + (size_t)node * D_OUT;
    float x0 = r[lane];
    float x1 = (lane < D_OUT - 32) ? r[32 + lane] : -INFINITY;
    float mx = fmaxf(x0, x1);
#pragma unroll
    for (int off = 16; off > 0; off >>= 1) mx = fmaxf(mx, __shfl_xor(mx, off, 32));
    float s = __expf(x0 - mx) + ((lane < D_OUT - 32) ? __expf(x1 - mx) : 0.0f);
#pragma unroll
    for (int off = 16; off > 0; off >>= 1) s += __shfl_xor(s, off, 32);
    float lse = mx + __logf(s);
    float* o = out + (size_t)node * D_OUT;
    o[lane] = x0 - lse;
    if (lane < D_OUT - 32) o[32 + lane] = x1 - lse;
}

// ---------------------------------------------------------------------------
// Launch
// ---------------------------------------------------------------------------
extern "C" void kernel_launch(void* const* d_in, const int* in_sizes, int n_in,
                              void* d_out, int out_size, void* d_ws, size_t ws_size,
                              hipStream_t stream)
{
    const float* x   = (const float*)d_in[0];
    const float* W1  = (const float*)d_in[1];
    const float* b1  = (const float*)d_in[2];
    const float* W2  = (const float*)d_in[3];
    const float* b2  = (const float*)d_in[4];
    const float* ew  = (const float*)d_in[5];
    const int*   es  = (const int*)d_in[6];
    const int*   ed  = (const int*)d_in[7];
    float*       out = (float*)d_out;

    // workspace layout (256B-aligned offsets)
    char* ws = (char*)d_ws;
    __bf16* w1t      = (__bf16*)(ws + 0);                    //     65,536 B
    __bf16* w2t      = (__bf16*)(ws + 65536);                //     12,288 B
    float*  support1 = (float*)(ws + 77824);                 // 51,200,000 B
    float*  agg1     = (float*)(ws + 77824 + 51200000ull);   // 51,200,000 B
    float*  support2 = (float*)(ws + 77824 + 102400000ull);  // 16,000,000 B
    float*  agg2     = (float*)(ws + 77824 + 118400000ull);  // 16,000,000 B

    const int grid_gemm    = (M_TILES + 7) / 8;              // 782
    const int grid_edges   = (N_EDGES * 32 + 255) / 256;     // 200000
    const int grid_zero1   = (N_NODES * D_HID / 4 + 255) / 256;
    const int grid_initb2  = (N_NODES * D_OUT + 255) / 256;
    const int grid_lsm     = (N_NODES * 32 + 255) / 256;

    gcn_prep_kernel<<<152, 256, 0, stream>>>(W1, W2, w1t, w2t);
    gcn_gemm1_kernel<<<grid_gemm, 256, 0, stream>>>(x, w1t, support1);
    gcn_zero_kernel<<<grid_zero1, 256, 0, stream>>>((float4*)agg1, N_NODES * D_HID / 4);
    gcn_scatter1_kernel<<<grid_edges, 256, 0, stream>>>(support1, es, ed, ew, agg1);
    gcn_gemm2_kernel<<<grid_gemm, 256, 0, stream>>>(agg1, b1, w2t, support2);
    gcn_initb2_kernel<<<grid_initb2, 256, 0, stream>>>(agg2, b2);
    gcn_scatter2_kernel<<<grid_edges, 256, 0, stream>>>(support2, es, ed, ew, agg2);
    gcn_lsm_kernel<<<grid_lsm, 256, 0, stream>>>(agg2, out);
}
